// Attention_50259707298691
// MI455X (gfx1250) — matrix operand order repaired and checked by reading, not compiled, not run on hardware
//
#include <hip/hip_runtime.h>

#define H 512
#define B 32
#define T 2048
#define M_TOTAL (B * T)   // 65536 rows, row r = t*B + b (contiguous in encoder_outputs)
#define WROW 1536         // W_attn row stride (3H)
#define W2OFF 512         // enc block offset inside W_attn rows
#define W3OFF 1024        // coverage block offset

#define AS1 __attribute__((address_space(1)))
#define AS3 __attribute__((address_space(3)))

#if defined(__has_builtin)
#  if __has_builtin(__builtin_amdgcn_global_load_async_to_lds_b128)
#    define HAVE_ASYNC_LDS 1
#  else
#    define HAVE_ASYNC_LDS 0
#  endif
#else
#  define HAVE_ASYNC_LDS 0
#endif

typedef __attribute__((ext_vector_type(16))) __bf16 bf16x16;
typedef __attribute__((ext_vector_type(8)))  float  f32x8;
typedef __attribute__((ext_vector_type(4)))  int    v4i;

struct Pack32 { uint4 lo; uint4 hi; };   // 32 bytes -> bit_cast to bf16x16

__device__ __forceinline__ unsigned short f2bf(float f) {
    unsigned int u = __builtin_bit_cast(unsigned int, f);
    u += 0x7FFFu + ((u >> 16) & 1u);      // round-to-nearest-even
    return (unsigned short)(u >> 16);
}

__device__ __forceinline__ float blockReduceSum256(float v, float* red) {
    int tid = threadIdx.x;
    red[tid] = v; __syncthreads();
    for (int s = 128; s > 0; s >>= 1) {
        if (tid < s) red[tid] += red[tid + s];
        __syncthreads();
    }
    float r = red[0]; __syncthreads();
    return r;
}

// ---------------- kernel 0: Ah[b,h] = hidden[b]·W1[h,:] + b_attn[h];  u[h] = W3[h,:]·W_cov
__global__ void prep_kernel(const float* __restrict__ hidden,
                            const float* __restrict__ W_attn,
                            const float* __restrict__ b_attn,
                            const float* __restrict__ W_cov,
                            float* __restrict__ Ah,
                            float* __restrict__ u) {
    __shared__ float w1[512];
    __shared__ float red[256];
    const int h = blockIdx.x, tid = threadIdx.x;
    const float* wr = W_attn + (size_t)h * WROW;
    w1[tid]       = wr[tid];
    w1[tid + 256] = wr[tid + 256];
    float up = wr[W3OFF + tid] * W_cov[tid] + wr[W3OFF + 256 + tid] * W_cov[256 + tid];
    __syncthreads();
    float ut = blockReduceSum256(up, red);
    if (tid == 0) u[h] = ut;
    for (int b = 0; b < B; ++b) {
        float p = w1[tid] * hidden[b * H + tid] + w1[tid + 256] * hidden[b * H + 256 + tid];
        float s = blockReduceSum256(p, red);
        if (tid == 0) Ah[b * H + h] = s + b_attn[h];
    }
}

// ---------------- kernel 1: W2 (f32) -> bf16, scattered into WMMA B-fragment order
// wsB[((kt*32 + nt)*32 + l)*16 + e] = bf16( W_attn[n, W2OFF + k] )
// n = nt*16 + (l&15);  k = kt*32 + ((l>=16)?16:0) + e   (contiguous k order, same as the A fragments)
__global__ void convB_kernel(const float* __restrict__ W_attn,
                             unsigned short* __restrict__ wsB) {
    int id = blockIdx.x * 256 + threadIdx.x;       // 262144 total
    int e  = id & 15;
    int l  = (id >> 4) & 31;
    int nt = (id >> 9) & 31;
    int kt = id >> 14;
    int n  = nt * 16 + (l & 15);
    int k  = kt * 32 + ((l >> 4) << 4) + e;
    wsB[id] = f2bf(W_attn[(size_t)n * WROW + W2OFF + k]);
}

// ---------------- kernel 2: main fused GEMM + relu + dot(v) epilogue
// Block = 512 threads = 16 waves. WG covers 64 rows x all 512 cols.
// Wave w owns N-tiles {2w, 2w+1} and all four 16-row M strips:
// B fragments are loaded ONCE per WG; A fragments reused across both N-tiles.
__global__ void __launch_bounds__(512) attn_main_kernel(
    const float* __restrict__ X,         // encoder_outputs, flat [T*B, H], row r = t*B+b
    const float* __restrict__ coverage,  // [B, T]
    const unsigned short* __restrict__ wsB,
    const float* __restrict__ Ah,        // [B, H]
    const float* __restrict__ u,         // [H]
    const float* __restrict__ vvec,      // [H]
    float* __restrict__ scores)          // [B, T]
{
    __shared__ __align__(16) unsigned short Xs[64 * 520];  // 66.5 KB bf16, padded stride
    __shared__ __align__(16) float AhS[B * H];             // 64 KB
    __shared__ float covS[64];
    __shared__ float scoreS[16][64];

    const int tid   = threadIdx.x;
    const int rbase = blockIdx.x * 64;

    // --- stage Ah[32][512] into LDS: raw 64KB copy, async global->LDS if available
#if HAVE_ASYNC_LDS
#pragma unroll
    for (int i = 0; i < 8; ++i) {
        int idx4 = tid + i * 512;                      // float4 index, 4096 total
        __builtin_amdgcn_global_load_async_to_lds_b128(
            (AS1 v4i*)(Ah + (size_t)idx4 * 4),
            (AS3 v4i*)&AhS[idx4 * 4], 0, 0);
    }
#else
    {
        const float4* s4 = (const float4*)Ah;
        float4*       d4 = (float4*)AhS;
#pragma unroll
        for (int i = 0; i < 8; ++i) d4[tid + i * 512] = s4[tid + i * 512];
    }
#endif

    if (tid < 64) {
        int r = rbase + tid;
        covS[tid] = coverage[(r & 31) * T + (r >> 5)];
    }
    {   // stage X tile: 64 rows x 512 f32 -> bf16 LDS (read-once from HBM)
        int row = tid >> 3;
        int kq  = (tid & 7) * 64;
        const float4* src = (const float4*)(X + (size_t)(rbase + row) * H + kq);
        unsigned short* dst = &Xs[row * 520 + kq];
#pragma unroll 4
        for (int i = 0; i < 16; ++i) {
            float4 f = src[i];
            ushort4 h4;
            h4.x = f2bf(f.x); h4.y = f2bf(f.y); h4.z = f2bf(f.z); h4.w = f2bf(f.w);
            *(ushort4*)(dst + 4 * i) = h4;
        }
    }
    __syncthreads();

    const int w  = tid >> 5;   // wave 0..15 -> owns ntiles 2w, 2w+1
    const int l  = tid & 31;
    const int lh = l & 15;
    const int hi = l >> 4;

    float un[2], vn[2];
#pragma unroll
    for (int j = 0; j < 2; ++j) {
        int n = (w * 2 + j) * 16 + lh;
        un[j] = u[n];
        vn[j] = vvec[n];
    }

    // 8 accumulators: acc[ms*2+j] for ms=0..3, j=0..1
    f32x8 acc[8];
#pragma unroll
    for (int q = 0; q < 8; ++q) acc[q] = (f32x8){0,0,0,0,0,0,0,0};

    const unsigned short* arow = &Xs[lh * 520 + (hi << 4)];  // + ms*16*520 + kt*32

#pragma unroll 1
    for (int kt = 0; kt < 16; ++kt) {
        // load the four A fragments for this K step (reused by both N-tiles)
        bf16x16 af[4];
#pragma unroll
        for (int ms = 0; ms < 4; ++ms) {
            const unsigned short* ap = arow + ms * (16 * 520) + kt * 32;
            Pack32 pa;
            pa.lo = *(const uint4*)(ap);
            pa.hi = *(const uint4*)(ap + 8);
            af[ms] = __builtin_bit_cast(bf16x16, pa);
        }
        if (kt + 1 < 16)
            __builtin_prefetch(wsB + (size_t)(((kt + 1) * 32 + w * 2) * 32 + l) * 16, 0, 1);
#pragma unroll
        for (int j = 0; j < 2; ++j) {
            const uint4* bp =
                (const uint4*)(wsB + (size_t)((kt * 32 + (w * 2 + j)) * 32 + l) * 16);
            Pack32 pb; pb.lo = bp[0]; pb.hi = bp[1];
            bf16x16 bf = __builtin_bit_cast(bf16x16, pb);
#pragma unroll
            for (int ms = 0; ms < 4; ++ms) {
                acc[ms * 2 + j] = __builtin_amdgcn_wmma_f32_16x16x32_bf16(
                    false, af[ms], false, bf, (short)0, acc[ms * 2 + j], false, false);
            }
        }
    }

    // make sure the async Ah copy has landed (all waves), then barrier
#if HAVE_ASYNC_LDS
#  if __has_builtin(__builtin_amdgcn_s_wait_asynccnt)
    __builtin_amdgcn_s_wait_asynccnt(0);
#  else
    asm volatile("s_wait_asynccnt 0x0" ::: "memory");
#  endif
#endif
    __syncthreads();

    // epilogue: energy = relu(E + Ah + cov*u); partial score += energy * v[n]
    float ps[4][8];
#pragma unroll
    for (int ms = 0; ms < 4; ++ms)
#pragma unroll
        for (int vv = 0; vv < 8; ++vv) ps[ms][vv] = 0.f;

#pragma unroll
    for (int j = 0; j < 2; ++j) {
        int n = (w * 2 + j) * 16 + lh;
#pragma unroll
        for (int ms = 0; ms < 4; ++ms) {
#pragma unroll
            for (int vv = 0; vv < 8; ++vv) {
                int m_local = ms * 16 + vv + (hi << 3);   // C layout: M = vv + 8*hi
                int b = m_local & 31;                     // rbase multiple of 64
                float a = AhS[b * H + n];
                float e = acc[ms * 2 + j][vv] + a + covS[m_local] * un[j];
                e = fmaxf(e, 0.f);
                ps[ms][vv] += e * vn[j];
            }
        }
    }
    // reduce over the 16 lanes of each half (they hold the 16 N columns of each tile)
#pragma unroll
    for (int ms = 0; ms < 4; ++ms) {
#pragma unroll
        for (int vv = 0; vv < 8; ++vv) {
            float s = ps[ms][vv];
            s += __shfl_xor(s, 1, 32);
            s += __shfl_xor(s, 2, 32);
            s += __shfl_xor(s, 4, 32);
            s += __shfl_xor(s, 8, 32);
            ps[ms][vv] = s;
        }
    }
    if (lh == 0) {
#pragma unroll
        for (int ms = 0; ms < 4; ++ms)
#pragma unroll
            for (int vv = 0; vv < 8; ++vv)
                scoreS[w][ms * 16 + (hi << 3) + vv] = ps[ms][vv];
    }
    __syncthreads();
    if (tid < 64) {
        float s = 0.f;
#pragma unroll
        for (int ww = 0; ww < 16; ++ww) s += scoreS[ww][tid];
        int r = rbase + tid;
        scores[(r & 31) * T + (r >> 5)] = s;
    }
}

// ---------------- kernel 3: softmax over T per batch + coverage update
__global__ void softmax_kernel(const float* __restrict__ scores,
                               const float* __restrict__ coverage,
                               float* __restrict__ out) {
    __shared__ float red[256];
    const int b = blockIdx.x, tid = threadIdx.x;
    float loc[8];
    float mx = -3.402823466e38f;
#pragma unroll
    for (int i = 0; i < 8; ++i) {
        float s = scores[b * T + tid + i * 256];
        loc[i] = s; mx = fmaxf(mx, s);
    }
    red[tid] = mx; __syncthreads();
    for (int s = 128; s > 0; s >>= 1) {
        if (tid < s) red[tid] = fmaxf(red[tid], red[tid + s]);
        __syncthreads();
    }
    float MX = red[0]; __syncthreads();
    float sum = 0.f;
#pragma unroll
    for (int i = 0; i < 8; ++i) { float e = __expf(loc[i] - MX); loc[i] = e; sum += e; }
    red[tid] = sum; __syncthreads();
    for (int s = 128; s > 0; s >>= 1) {
        if (tid < s) red[tid] += red[tid + s];
        __syncthreads();
    }
    float rinv = 1.f / red[0];
#pragma unroll
    for (int i = 0; i < 8; ++i) {
        int t = tid + i * 256;
        float a = loc[i] * rinv;
        out[b * T + t] = a;                               // attn_weights [B,1,T]
        out[B * T + b * T + t] = coverage[b * T + t] + a; // coverage_new [B,T]
    }
}

extern "C" void kernel_launch(void* const* d_in, const int* in_sizes, int n_in,
                              void* d_out, int out_size, void* d_ws, size_t ws_size,
                              hipStream_t stream) {
    const float* hidden = (const float*)d_in[0];
    const float* enc    = (const float*)d_in[1];
    const float* cov    = (const float*)d_in[2];
    const float* W_attn = (const float*)d_in[3];
    const float* b_attn = (const float*)d_in[4];
    const float* v      = (const float*)d_in[5];
    const float* W_cov  = (const float*)d_in[6];
    float* out = (float*)d_out;

    char* ws = (char*)d_ws;
    unsigned short* wsB = (unsigned short*)(ws);               // 512 KB bf16 B-frags
    float* Ah     = (float*)(ws + 524288);                     // 64 KB
    float* u      = (float*)(ws + 524288 + 65536);             // 2 KB (pad to 4 KB)
    float* scores = (float*)(ws + 524288 + 65536 + 4096);      // 256 KB

    prep_kernel   <<<H, 256, 0, stream>>>(hidden, W_attn, b_attn, W_cov, Ah, u);
    convB_kernel  <<<(512 * 512) / 256, 256, 0, stream>>>(W_attn, wsB);
    attn_main_kernel<<<M_TOTAL / 64, 512, 0, stream>>>(enc, cov, wsB, Ah, u, v, scores);
    softmax_kernel<<<B, 256, 0, stream>>>(scores, cov, out);
}
